// Model_NIH_59476707115580
// MI455X (gfx1250) — matrix operand order repaired and checked by reading, not compiled, not run on hardware
//
#include <hip/hip_runtime.h>
#include <hip/hip_bf16.h>

typedef __attribute__((ext_vector_type(16))) _Float16 v16h;
typedef __attribute__((ext_vector_type(8)))  float    v8f;

#define EPSBN 1e-5f
#define TBM 64
#define TBN 64
#define TBK 32

// ---------------------------------------------------------------------------
// Weight prep: fold BN (scale = g*rsqrt(v+eps)) into fp16 weights + fp32 bias
// ---------------------------------------------------------------------------
__global__ void k_fold_bn(const float* __restrict__ w, const float* __restrict__ g,
                          const float* __restrict__ b, const float* __restrict__ m,
                          const float* __restrict__ v, _Float16* __restrict__ wo,
                          float* __restrict__ bo, int Cout, int K) {
  int i = blockIdx.x * blockDim.x + threadIdx.x;
  int total = Cout * K;
  if (i < total) {
    int oc = i / K;
    float s = g[oc] * rsqrtf(v[oc] + EPSBN);
    wo[i] = (_Float16)(w[i] * s);
  }
  if (i < Cout) {
    float s = g[i] * rsqrtf(v[i] + EPSBN);
    bo[i] = b[i] - m[i] * s;
  }
}

__global__ void k_f2h(const float* __restrict__ w, _Float16* __restrict__ o, int n) {
  int i = blockIdx.x * blockDim.x + threadIdx.x;
  if (i < n) o[i] = (_Float16)w[i];
}

// ---------------------------------------------------------------------------
// Fragment-major LDS layout (CDNA5 ISA 05_wmma):
// A 16x32 f16:  lane = (m&15) + 16*((k>>3)&1), elem j = ((k>>4)&1)*8 + (k&7)
// B 32x16 f16:  lane = (n&15) + 16*((k>>3)&1), elem j = ((k>>4)&1)*8 + (k&7)
// C/D 16x16 f32: lane holds col N=(lane&15); VGPR r holds row M = r + 8*(lane>>4)
// Block tile 64x64x32, 8 waves; wave = (msub, npair); each wave does 2 WMMAs
// per K-step sharing one A fragment.
// ---------------------------------------------------------------------------

__global__ __launch_bounds__(256) void k_conv_wmma(
    const _Float16* __restrict__ Wh, const float* __restrict__ X,
    const float* __restrict__ bias, const float* __restrict__ resid,
    float* __restrict__ Y,
    int Bn, int Cin, int IH, int IW, int Cout, int OH, int OW,
    int KH, int KW, int stride, int pad, int relu)
{
  __shared__ __align__(32) _Float16 ldsA[TBM * TBK];  // 4 subtiles x 32 lanes x 16
  __shared__ __align__(32) _Float16 ldsB[TBK * TBN];  // 4 subtiles x 32 lanes x 16

  const int tid   = threadIdx.x;
  const int lane  = tid & 31;
  const int wave  = tid >> 5;
  const int msub  = wave & 3;
  const int nsub0 = (wave >> 2) << 1;

  const int bN   = blockIdx.x * TBN;
  const int bM   = blockIdx.y * TBM;
  const int Npix = Bn * OH * OW;
  const int Ktot = Cin * KH * KW;
  const int KK   = KH * KW;

  // K-invariant staging constants
  const int a_kl  = tid & 31;                       // A: k within tile (fixed)
  const int a_j   = ((a_kl >> 4) & 1) * 8 + (a_kl & 7);
  const int a_lhi = (a_kl & 8) ? 16 : 0;
  const int a_ml0 = tid >> 5;                       // ml = a_ml0 + t*8

  const int b_nl  = tid & 63;                       // B: n within tile (fixed)
  const int b_kl0 = tid >> 6;                       // kl = b_kl0 + t*4
  const int b_sub = (b_nl >> 4) << 9;
  const int b_l   = b_nl & 15;
  const int np    = bN + b_nl;
  const bool npv  = np < Npix;
  int pb = 0, poy = 0, pox = 0;
  if (npv) {
    pb = np / (OH * OW);
    int rem = np - pb * OH * OW;
    poy = rem / OW;
    pox = rem - poy * OW;
  }
  const float* Xb = X + (long long)pb * Cin * IH * IW;
  const int oys = poy * stride - pad;
  const int oxs = pox * stride - pad;

  v8f acc0 = {}, acc1 = {};

  for (int k0 = 0; k0 < Ktot; k0 += TBK) {
    // stage A (fp16 weights)
#pragma unroll
    for (int t = 0; t < 8; ++t) {
      int ml = a_ml0 + t * 8;
      int oc = bM + ml;
      int gk = k0 + a_kl;
      _Float16 val = (_Float16)0.f;
      if (oc < Cout && gk < Ktot) val = Wh[(long long)oc * Ktot + gk];
      ldsA[((ml >> 4) << 9) + (((ml & 15) + a_lhi) << 4) + a_j] = val;
    }
    // prefetch next K-step of the weight stream (global_prefetch_b8)
    if (k0 + TBK < Ktot) {
      int oc = bM + a_ml0;
      if (oc < Cout)
        __builtin_prefetch(&Wh[(long long)oc * Ktot + k0 + TBK + a_kl], 0, 0);
    }
    // stage B (im2col of fp32 input -> fp16)
#pragma unroll
    for (int t = 0; t < 8; ++t) {
      int kl = b_kl0 + t * 4;
      int gk = k0 + kl;
      _Float16 val = (_Float16)0.f;
      if (npv && gk < Ktot) {
        int ic = gk / KK;
        int r  = gk - ic * KK;
        int ky = r / KW;
        int kx = r - ky * KW;
        int iy = oys + ky, ix = oxs + kx;
        if (iy >= 0 && iy < IH && ix >= 0 && ix < IW)
          val = (_Float16)Xb[((long long)ic * IH + iy) * IW + ix];
      }
      ldsB[b_sub + ((b_l + ((kl & 8) ? 16 : 0)) << 4) + (((kl >> 4) & 1) * 8 + (kl & 7))] = val;
    }
    __syncthreads();
    v16h a   = *(const v16h*)&ldsA[(msub << 9) + (lane << 4)];
    v16h bf0 = *(const v16h*)&ldsB[(nsub0 << 9) + (lane << 4)];
    v16h bf1 = *(const v16h*)&ldsB[((nsub0 + 1) << 9) + (lane << 4)];
    acc0 = __builtin_amdgcn_wmma_f32_16x16x32_f16(false, a, false, bf0,
                                                  (short)0, acc0, false, false);
    acc1 = __builtin_amdgcn_wmma_f32_16x16x32_f16(false, a, false, bf1,
                                                  (short)0, acc1, false, false);
    __syncthreads();
  }

  // epilogue: bias + residual + relu, fp32 NCHW
  auto epi = [&](const v8f& acc, int nsub) {
    int ncol = bN + (nsub << 4) + (lane & 15);
    if (ncol < Npix) {
      int b   = ncol / (OH * OW);
      int rem = ncol - b * OH * OW;
      int oy  = rem / OW;
      int ox  = rem - oy * OW;
#pragma unroll
      for (int r = 0; r < 8; ++r) {
        int oc = bM + (msub << 4) + r + ((lane >> 4) << 3);
        if (oc < Cout) {
          float val = acc[r];
          if (bias)  val += bias[oc];
          long long off = (((long long)b * Cout + oc) * OH + oy) * OW + ox;
          if (resid) val += resid[off];
          if (relu)  val = fmaxf(val, 0.f);
          Y[off] = val;
        }
      }
    }
  };
  epi(acc0, nsub0);
  epi(acc1, nsub0 + 1);
}

// Batched fp32-in/fp32-out GEMM via fp16 WMMA: C[z] = op(A[z]) * op(B[z])
__global__ __launch_bounds__(256) void k_gemm_wmma(
    const float* __restrict__ Ag, const float* __restrict__ Bg, float* __restrict__ Cg,
    int M, int N, int K, int transA, int transB,
    long long sA, long long sB, long long sC, int lda, int ldb)
{
  __shared__ __align__(32) _Float16 ldsA[TBM * TBK];
  __shared__ __align__(32) _Float16 ldsB[TBK * TBN];

  const float* Ab = Ag + (long long)blockIdx.z * sA;
  const float* Bb = Bg + (long long)blockIdx.z * sB;
  float*       Cb = Cg + (long long)blockIdx.z * sC;

  const int tid   = threadIdx.x;
  const int lane  = tid & 31;
  const int wave  = tid >> 5;
  const int msub  = wave & 3;
  const int nsub0 = (wave >> 2) << 1;
  const int bM    = blockIdx.y * TBM;
  const int bN    = blockIdx.x * TBN;

  const int a_kl  = tid & 31;
  const int a_j   = ((a_kl >> 4) & 1) * 8 + (a_kl & 7);
  const int a_lhi = (a_kl & 8) ? 16 : 0;
  const int a_ml0 = tid >> 5;

  const int b_nl  = tid & 63;
  const int b_kl0 = tid >> 6;
  const int b_sub = (b_nl >> 4) << 9;
  const int b_l   = b_nl & 15;
  const int gn_st = bN + b_nl;

  v8f acc0 = {}, acc1 = {};

  for (int k0 = 0; k0 < K; k0 += TBK) {
#pragma unroll
    for (int t = 0; t < 8; ++t) {
      int ml = a_ml0 + t * 8;
      int gm = bM + ml, gk = k0 + a_kl;
      _Float16 val = (_Float16)0.f;
      if (gm < M && gk < K)
        val = (_Float16)(transA ? Ab[(long long)gk * lda + gm]
                                : Ab[(long long)gm * lda + gk]);
      ldsA[((ml >> 4) << 9) + (((ml & 15) + a_lhi) << 4) + a_j] = val;
    }
#pragma unroll
    for (int t = 0; t < 8; ++t) {
      int kl = b_kl0 + t * 4;
      int gk = k0 + kl;
      _Float16 val = (_Float16)0.f;
      if (gn_st < N && gk < K)
        val = (_Float16)(transB ? Bb[(long long)gn_st * ldb + gk]
                                : Bb[(long long)gk * ldb + gn_st]);
      ldsB[b_sub + ((b_l + ((kl & 8) ? 16 : 0)) << 4) + (((kl >> 4) & 1) * 8 + (kl & 7))] = val;
    }
    __syncthreads();
    v16h a   = *(const v16h*)&ldsA[(msub << 9) + (lane << 4)];
    v16h bf0 = *(const v16h*)&ldsB[(nsub0 << 9) + (lane << 4)];
    v16h bf1 = *(const v16h*)&ldsB[((nsub0 + 1) << 9) + (lane << 4)];
    acc0 = __builtin_amdgcn_wmma_f32_16x16x32_f16(false, a, false, bf0,
                                                  (short)0, acc0, false, false);
    acc1 = __builtin_amdgcn_wmma_f32_16x16x32_f16(false, a, false, bf1,
                                                  (short)0, acc1, false, false);
    __syncthreads();
  }

  auto epi = [&](const v8f& acc, int nsub) {
    int gn = bN + (nsub << 4) + (lane & 15);
    if (gn < N) {
#pragma unroll
      for (int r = 0; r < 8; ++r) {
        int gm = bM + (msub << 4) + r + ((lane >> 4) << 3);
        if (gm < M) Cb[(long long)gm * N + gn] = acc[r];
      }
    }
  };
  epi(acc0, nsub0);
  epi(acc1, nsub0 + 1);
}

// 3x3 s2 pad1 maxpool, -inf padding (matches jax reduce_window)
__global__ void k_maxpool(const float* __restrict__ X, float* __restrict__ Y,
                          int BC, int IH, int IW, int OH, int OW) {
  int i = blockIdx.x * blockDim.x + threadIdx.x;
  int total = BC * OH * OW;
  if (i >= total) return;
  int bc  = i / (OH * OW);
  int rem = i - bc * OH * OW;
  int oy  = rem / OW, ox = rem - oy * OW;
  float mx = -__builtin_inff();
  for (int ky = 0; ky < 3; ++ky) {
    int iy = oy * 2 - 1 + ky;
    if (iy < 0 || iy >= IH) continue;
    for (int kx = 0; kx < 3; ++kx) {
      int ix = ox * 2 - 1 + kx;
      if (ix < 0 || ix >= IW) continue;
      mx = fmaxf(mx, X[((long long)bc * IH + iy) * IW + ix]);
    }
  }
  Y[i] = mx;
}

// row softmax, 256 cols, one wave per row, 8 rows per block
__global__ void k_softmax256(float* __restrict__ S, int rows) {
  int wv = threadIdx.x >> 5, lane = threadIdx.x & 31;
  int row = blockIdx.x * 8 + wv;
  if (row >= rows) return;
  float* p = S + (long long)row * 256;
  float v[8];
  float mx = -__builtin_inff();
#pragma unroll
  for (int i = 0; i < 8; ++i) { v[i] = p[lane + (i << 5)]; mx = fmaxf(mx, v[i]); }
  for (int o = 16; o; o >>= 1) mx = fmaxf(mx, __shfl_xor(mx, o, 32));
  float sum = 0.f;
#pragma unroll
  for (int i = 0; i < 8; ++i) { v[i] = expf(v[i] - mx); sum += v[i]; }
  for (int o = 16; o; o >>= 1) sum += __shfl_xor(sum, o, 32);
  float inv = 1.f / sum;
#pragma unroll
  for (int i = 0; i < 8; ++i) p[lane + (i << 5)] = v[i] * inv;
}

__global__ void k_avgpool(const float* __restrict__ X, float* __restrict__ F,
                          int BC, int HW) {
  int i = blockIdx.x * blockDim.x + threadIdx.x;
  if (i < BC) {
    float s = 0.f;
    for (int j = 0; j < HW; ++j) s += X[(long long)i * HW + j];
    F[i] = s * (1.0f / HW);
  }
}

// y[n,b,c] = dot(feat[b], cls_w[n,c]) + cls_b[n,c]; one wave per output
__global__ void k_cls(const float* __restrict__ feat, const float* __restrict__ W,
                      const float* __restrict__ Bv, float* __restrict__ Y) {
  int gw   = (blockIdx.x * blockDim.x + threadIdx.x) >> 5;
  int lane = threadIdx.x & 31;
  if (gw >= 14 * 32 * 2) return;
  int n = gw / 64, r = gw - n * 64, b = r >> 1, c = r & 1;
  float s = 0.f;
  for (int d = lane; d < 512; d += 32) s += feat[b * 512 + d] * W[(n * 2 + c) * 512 + d];
  for (int o = 16; o; o >>= 1) s += __shfl_xor(s, o, 32);
  if (lane == 0) Y[gw] = s + Bv[n * 2 + c];
}

// ---------------------------------------------------------------------------
// Host orchestration
// ---------------------------------------------------------------------------
struct ConvW { const _Float16* w; const float* bias; };

static inline size_t al256(size_t x) { return (x + 255) & ~(size_t)255; }

static ConvW prep_bn(const float* w, const float* g, const float* b,
                     const float* m, const float* v, int Cout, int K,
                     char*& wa, char*& ba, hipStream_t s) {
  _Float16* wo = (_Float16*)wa; wa += al256((size_t)Cout * K * sizeof(_Float16));
  float*    bo = (float*)ba;    ba += al256((size_t)Cout * sizeof(float));
  int total = Cout * K;
  k_fold_bn<<<(total + 255) / 256, 256, 0, s>>>(w, g, b, m, v, wo, bo, Cout, K);
  return ConvW{wo, bo};
}

static const _Float16* prep_plain(const float* w, int n, char*& wa, hipStream_t s) {
  _Float16* wo = (_Float16*)wa; wa += al256((size_t)n * sizeof(_Float16));
  k_f2h<<<(n + 255) / 256, 256, 0, s>>>(w, wo, n);
  return wo;
}

static void conv(ConvW cw, const float* X, const float* resid, float* Y,
                 int Bn, int Cin, int IH, int IW, int Cout, int OH, int OW,
                 int KH, int KW, int stride, int pad, int relu, hipStream_t s) {
  int Npix = Bn * OH * OW;
  dim3 grid((Npix + TBN - 1) / TBN, (Cout + TBM - 1) / TBM, 1);
  k_conv_wmma<<<grid, 256, 0, s>>>(cw.w, X, cw.bias, resid, Y, Bn, Cin, IH, IW,
                                   Cout, OH, OW, KH, KW, stride, pad, relu);
}

extern "C" void kernel_launch(void* const* d_in, const int* in_sizes, int n_in,
                              void* d_out, int out_size, void* d_ws, size_t ws_size,
                              hipStream_t stream) {
  (void)in_sizes; (void)n_in; (void)out_size; (void)ws_size;
  int ii = 0;
  auto nip = [&]() -> const float* { return (const float*)d_in[ii++]; };

  // workspace layout
  char* wa = (char*)d_ws;                         // fp16 weight arena (36 MB)
  char* ba = (char*)d_ws + ((size_t)36 << 20);    // fp32 bias arena (4 MB)
  float* A  = (float*)((char*)d_ws + ((size_t)40 << 20)); // 33,554,432 floats
  float* Bf = A + 33554432;                       // 8,388,608 floats, separate region
  // carved from A once the stem output is dead:
  float* C = A;
  float* D = A + 8388608;
  float* E = A + 16777216;
  float* F = A + 25165824;

  const int Bn = 32;

  // ---- stem: conv7x7 s2 p3 + BN + relu, then maxpool 3x3 s2 p1 ----
  const float* x  = nip();
  const float* sw = nip();
  const float *sg = nip(), *sb = nip(), *sm = nip(), *sv = nip();
  ConvW cstem = prep_bn(sw, sg, sb, sm, sv, 64, 3 * 49, wa, ba, stream);
  conv(cstem, x, nullptr, A, Bn, 3, 256, 256, 64, 128, 128, 7, 7, 2, 3, 1, stream);
  {
    int total = Bn * 64 * 64 * 64;
    k_maxpool<<<(total + 255) / 256, 256, 0, stream>>>(A, Bf, Bn * 64, 128, 128, 64, 64);
  }

  // ---- basic block: relu(bn2(conv2(relu(bn1(conv1 x)))) + shortcut) ----
  auto run_block = [&](const float* xin, float* tmp1, float* sc, float* out,
                       int Cin, int H, int W, int Cout, int stride) {
    const float* w1 = nip();
    const float *g1 = nip(), *b1 = nip(), *m1 = nip(), *v1 = nip();
    const float* w2 = nip();
    const float *g2 = nip(), *b2 = nip(), *m2 = nip(), *v2 = nip();
    bool down = (stride != 1) || (Cin != Cout);
    const float *wd = nullptr, *gd = nullptr, *bd = nullptr, *md = nullptr, *vd = nullptr;
    if (down) { wd = nip(); gd = nip(); bd = nip(); md = nip(); vd = nip(); }
    ConvW c1 = prep_bn(w1, g1, b1, m1, v1, Cout, Cin * 9, wa, ba, stream);
    ConvW c2 = prep_bn(w2, g2, b2, m2, v2, Cout, Cout * 9, wa, ba, stream);
    int OH = (H + 2 - 3) / stride + 1, OW = (W + 2 - 3) / stride + 1;
    conv(c1, xin, nullptr, tmp1, Bn, Cin, H, W, Cout, OH, OW, 3, 3, stride, 1, 1, stream);
    const float* scp = xin;
    if (down) {
      ConvW cd = prep_bn(wd, gd, bd, md, vd, Cout, Cin, wa, ba, stream);
      conv(cd, xin, nullptr, sc, Bn, Cin, H, W, Cout, OH, OW, 1, 1, stride, 0, 0, stream);
      scp = sc;
    }
    conv(c2, tmp1, scp, out, Bn, Cout, OH, OW, Cout, OH, OW, 3, 3, 1, 1, 1, stream);
  };

  run_block(Bf, C, E, D, 64, 64, 64, 64, 1);     // L1B0 -> D
  run_block(D,  C, E, Bf, 64, 64, 64, 64, 1);    // L1B1 -> Bf
  run_block(Bf, C, D, E, 64, 64, 64, 128, 2);    // L2B0 -> E
  run_block(E,  C, D, Bf, 128, 32, 32, 128, 1);  // L2B1 -> Bf
  run_block(Bf, C, D, E, 128, 32, 32, 256, 2);   // L3B0 -> E
  run_block(E,  C, D, Bf, 256, 16, 16, 256, 1);  // L3B1 -> Bf   (x3 = Bf)

  // ---- cross attention at 16x16 (HW = 256) ----
  const float* wth = nip();
  const float* wph = nip();
  const float* wg  = nip();
  const float* wwz = nip();
  const _Float16* hth = prep_plain(wth, 128 * 256, wa, stream);
  const _Float16* hph = prep_plain(wph, 128 * 256, wa, stream);
  const _Float16* hg  = prep_plain(wg,  256 * 256, wa, stream);
  const _Float16* hwz = prep_plain(wwz, 256 * 256, wa, stream);

  conv(ConvW{hth, nullptr}, Bf, nullptr, C, Bn, 256, 16, 16, 128, 16, 16, 1, 1, 1, 0, 0, stream); // theta -> C [b,128,256]
  conv(ConvW{hph, nullptr}, Bf, nullptr, D, Bn, 256, 16, 16, 128, 16, 16, 1, 1, 1, 0, 0, stream); // phi   -> D
  conv(ConvW{hg,  nullptr}, Bf, nullptr, E, Bn, 256, 16, 16, 256, 16, 16, 1, 1, 1, 0, 0, stream); // g     -> E

  dim3 gatt(256 / TBN, 256 / TBM, Bn);
  // score[b,q,k] = sum_c theta[b,c,q] * phi[b,c,k]   (A^T * B)
  k_gemm_wmma<<<gatt, 256, 0, stream>>>(C, D, F, 256, 256, 128, 1, 0,
                                        32768, 32768, 65536, 256, 256);
  k_softmax256<<<(Bn * 256) / 8, 256, 0, stream>>>(F, Bn * 256);
  // outchw[b,c,q] = sum_k g[b,c,k] * attn[b,q,k]     (A * B^T)
  k_gemm_wmma<<<gatt, 256, 0, stream>>>(E, F, C, 256, 256, 256, 0, 1,
                                        65536, 65536, 65536, 256, 256);
  // wz 1x1 conv, fused h2 + x3 residual (no relu)
  conv(ConvW{hwz, nullptr}, C, Bf, D, Bn, 256, 16, 16, 256, 16, 16, 1, 1, 1, 0, 0, stream);

  // ---- layer4 ----
  run_block(D,  C, E, Bf, 256, 16, 16, 512, 2);  // L4B0 -> Bf
  run_block(Bf, C, E, D, 512, 8, 8, 512, 1);     // L4B1 -> D   (x4 = D)

  // ---- outputs: x4, feat, y ----
  float* out_x4   = (float*)d_out;
  float* out_feat = out_x4 + 1048576;   // 32*512*8*8
  float* out_y    = out_feat + 16384;   // 32*512
  hipMemcpyAsync(out_x4, D, (size_t)1048576 * sizeof(float),
                 hipMemcpyDeviceToDevice, stream);
  k_avgpool<<<(16384 + 255) / 256, 256, 0, stream>>>(D, out_feat, 32 * 512, 64);
  const float* cls_w = nip();
  const float* cls_b = nip();
  k_cls<<<112, 256, 0, stream>>>(out_feat, cls_w, cls_b, out_y);
}
